// HiggsAudioTransformer_82781199663130
// MI455X (gfx1250) — compile-verified, matrix operand order checked by reading, not run on hardware
//
#include <hip/hip_runtime.h>
#include <hip/hip_bf16.h>
#include <math.h>

#define TEXT_VOCAB 32000
#define CBK 1024
#define NCB 8
#define HDIM 2048
#define NH 16
#define NKV 4
#define HD 128
#define FF 8192
#define NLAYERS 2
#define SEQ 2048
#define EPSF 1e-5f

typedef __attribute__((ext_vector_type(16))) __bf16 v16bf;
typedef __attribute__((ext_vector_type(8)))  float v8f;

__device__ __forceinline__ __bf16 f2bf(float f) { return (__bf16)f; }

__device__ __forceinline__ float4 f4s(float4 a, float s) {
    a.x *= s; a.y *= s; a.z *= s; a.w *= s; return a;
}

// Convert 4x float4 (16 source floats in fragment order, two 8-float runs) into v16bf.
__device__ __forceinline__ v16bf cvt16(float4 a0, float4 a1, float4 a2, float4 a3) {
    v16bf d;
    d[0]  = f2bf(a0.x); d[1]  = f2bf(a0.y); d[2]  = f2bf(a0.z); d[3]  = f2bf(a0.w);
    d[4]  = f2bf(a1.x); d[5]  = f2bf(a1.y); d[6]  = f2bf(a1.z); d[7]  = f2bf(a1.w);
    d[8]  = f2bf(a2.x); d[9]  = f2bf(a2.y); d[10] = f2bf(a2.z); d[11] = f2bf(a2.w);
    d[12] = f2bf(a3.x); d[13] = f2bf(a3.y); d[14] = f2bf(a3.z); d[15] = f2bf(a3.w);
    return d;
}

// Scatter one A-staging strip (16 contiguous k at row amm of mtile amt) into
// fragment-order LDS: element e -> k = h*8 + e + (e>=8?8:0), lane = h*16+row.
__device__ __forceinline__ void scatterA(__bf16* lAbuf, int akof, int amt, int amm,
                                         float4 a0, float4 a1, float4 a2, float4 a3) {
    float fa[16] = { a0.x, a0.y, a0.z, a0.w, a1.x, a1.y, a1.z, a1.w,
                     a2.x, a2.y, a2.z, a2.w, a3.x, a3.y, a3.z, a3.w };
    #pragma unroll
    for (int i = 0; i < 16; ++i) {
        int kk = akof + i;
        int h  = (kk >> 3) & 1;
        int e  = (kk & 7) + ((kk >> 4) << 3);
        lAbuf[((amt * 32 + h * 16 + amm) << 4) + e] = f2bf(fa[i]);
    }
}

// Scatter one B-staging strip (16 contiguous n at k-row) into fragment order;
// dst is pre-offset by ((ntile*32 + h*16)<<4) + e, stride 32B per n.
__device__ __forceinline__ void scatterB(__bf16* dst,
                                         float4 b0, float4 b1, float4 b2, float4 b3) {
    float fb[16] = { b0.x, b0.y, b0.z, b0.w, b1.x, b1.y, b1.z, b1.w,
                     b2.x, b2.y, b2.z, b2.w, b3.x, b3.y, b3.z, b3.w };
    #pragma unroll
    for (int i = 0; i < 16; ++i) dst[i << 4] = f2bf(fb[i]);
}

// ---------------------------------------------------------------------------
// GEMM v3: C[M,N] = A[M,K] x B[K,N], fp32 in/out, bf16 WMMA 16x16x32 compute.
// Block tile 128x128, K-step 32, LDS DOUBLE-BUFFERED (one barrier per step):
// issue next tile's global loads -> 8 WMMAs from current LDS buffer ->
// cvt+scatter next tile into alternate buffer -> s_barrier. 8 waves as
// 4(M)x2(N); each wave owns 2x4 16x16 tiles. Grid = (N/128, M/128).
// ---------------------------------------------------------------------------
__global__ __launch_bounds__(256)
void gemm_bf16_wmma(const float* __restrict__ A, const float* __restrict__ B,
                    float* __restrict__ C, int M, int N, int K) {
    __shared__ __bf16 lA[2][4096];   // [buf][mtile(8)][lane(32)][e(16)]  2x8KB
    __shared__ __bf16 lB[2][4096];   // [buf][ntile(8)][lane(32)][e(16)]  2x8KB

    const int t    = threadIdx.x;
    const int lane = t & 31;
    const int wave = t >> 5;
    const int half = lane >> 4;
    const int mr   = lane & 15;
    const int mw   = wave >> 1;       // 0..3
    const int nw   = wave & 1;        // 0..1
    const int blockN = blockIdx.x * 128;
    const int blockM = blockIdx.y * 128;

    // Staging roles (constant per thread)
    const int am   = t >> 1;               // A: row 0..127
    const int akof = (t & 1) * 16;         // A: k-offset 0 or 16
    const int amt  = am >> 4, amm = am & 15;
    const int bkk  = t >> 3;               // B: k row 0..31
    const int bng  = (t & 7) * 16;         // B: n-group base
    const int bh   = (bkk >> 3) & 1;
    const int be   = (bkk & 7) + ((bkk >> 4) << 3);
    const int bofs = (((t & 7) * 32 + bh * 16) << 4) + be;

    const float* pabase = A + (size_t)(blockM + am) * K + akof;
    const float* pbbase = B + (size_t)bkk * N + blockN + bng;

    v8f acc[2][4] = {};

    // ---- prologue: stage first K-step into buffer 0 ----
    {
        const float* pa = pabase;
        const float* pb = pbbase;
        scatterA(lA[0], akof, amt, amm,
                 *(const float4*)(pa),     *(const float4*)(pa + 4),
                 *(const float4*)(pa + 8), *(const float4*)(pa + 12));
        scatterB(lB[0] + bofs,
                 *(const float4*)(pb),     *(const float4*)(pb + 4),
                 *(const float4*)(pb + 8), *(const float4*)(pb + 12));
    }
    __syncthreads();

    int cur = 0;
    for (int k0 = 0; k0 < K; k0 += 32) {
        const int  nxt     = cur ^ 1;
        const bool hasNext = (k0 + 32) < K;

        // 1) issue next tile's global loads (latency hidden behind WMMAs)
        float4 a0, a1, a2, a3, b0, b1, b2, b3;
        if (hasNext) {
            const float* pa = pabase + (k0 + 32);
            const float* pb = pbbase + (size_t)(k0 + 32) * N;
            __builtin_prefetch(pa + 32, 0, 1);                 // k0+64 hint
            __builtin_prefetch(pb + (size_t)32 * N, 0, 1);
            a0 = *(const float4*)(pa);      a1 = *(const float4*)(pa + 4);
            a2 = *(const float4*)(pa + 8);  a3 = *(const float4*)(pa + 12);
            b0 = *(const float4*)(pb);      b1 = *(const float4*)(pb + 4);
            b2 = *(const float4*)(pb + 8);  b3 = *(const float4*)(pb + 12);
        }

        // 2) compute from current buffer
        v16bf af0 = *(const v16bf*)(&lA[cur][((mw * 2 + 0) * 32 + lane) << 4]);
        v16bf af1 = *(const v16bf*)(&lA[cur][((mw * 2 + 1) * 32 + lane) << 4]);
        #pragma unroll
        for (int j = 0; j < 4; ++j) {
            v16bf bfj = *(const v16bf*)(&lB[cur][((nw * 4 + j) * 32 + lane) << 4]);
            acc[0][j] = __builtin_amdgcn_wmma_f32_16x16x32_bf16(
                false, af0, false, bfj, (short)0, acc[0][j], false, false);
            acc[1][j] = __builtin_amdgcn_wmma_f32_16x16x32_bf16(
                false, af1, false, bfj, (short)0, acc[1][j], false, false);
        }

        // 3) convert + scatter next tile into alternate buffer
        if (hasNext) {
            scatterA(lA[nxt], akof, amt, amm, a0, a1, a2, a3);
            scatterB(lB[nxt] + bofs, b0, b1, b2, b3);
        }

        // 4) single barrier: orders store->load (nxt) and load->overwrite (cur)
        __syncthreads();
        cur = nxt;
    }

    #pragma unroll
    for (int i = 0; i < 2; ++i) {
        int rowbase = blockM + (mw * 2 + i) * 16 + 8 * half;
        #pragma unroll
        for (int j = 0; j < 4; ++j) {
            float* cp = C + (size_t)rowbase * N + blockN + (nw * 4 + j) * 16 + mr;
            #pragma unroll
            for (int r = 0; r < 8; ++r) cp[(size_t)r * N] = acc[i][j][r];
        }
    }
}

// ---------------------------------------------------------------------------
// WMMA flash attention (causal, GQA). Block = one head x 128 queries
// (8 waves x 16-query tiles). Loop over 32-key blocks: stage K/V tiles into
// LDS in B-fragment order, QK^T = 8 WMMAs, online softmax (row reductions via
// 16-lane shfl), P -> LDS in A-fragment order, PV = 8 WMMAs.
// Grid = (SEQ/128, NH), block = 256.
// ---------------------------------------------------------------------------
__global__ __launch_bounds__(256)
void flash_attn_wmma(const float* __restrict__ q, const float* __restrict__ kbuf,
                     const float* __restrict__ vbuf, float* __restrict__ out) {
    __shared__ __bf16 lK[4096];     // [ks(4)][nt(2)][lane(32)][e(16)]  8KB
    __shared__ __bf16 lV[4096];     // [nt(8)][lane(32)][e(16)]         8KB
    __shared__ __bf16 lP[8][512];   // per-wave 16x32 A-fragment        8KB

    const int tid  = threadIdx.x;
    const int wave = tid >> 5;
    const int lane = tid & 31;
    const int half = lane >> 4;
    const int mr   = lane & 15;
    const int qbase = blockIdx.x * 128;
    const int head  = blockIdx.y;
    const int kvh   = head / (NH / NKV);
    const int qrow  = qbase + wave * 16 + mr;
    const float scale = 0.08838834764831845f;   // 1/sqrt(128)

    // Q fragments, prescaled by 1/sqrt(HD)
    v16bf qf[4];
    {
        const float* qp = q + ((size_t)qrow * NH + head) * HD;
        #pragma unroll
        for (int ks = 0; ks < 4; ++ks) {
            const float* p = qp + ks * 32 + half * 8;
            qf[ks] = cvt16(f4s(*(const float4*)(p),      scale),
                           f4s(*(const float4*)(p + 4),  scale),
                           f4s(*(const float4*)(p + 16), scale),
                           f4s(*(const float4*)(p + 20), scale));
        }
    }

    v8f oacc[8] = {};
    float m_run[8], l_run[8];
    #pragma unroll
    for (int r = 0; r < 8; ++r) { m_run[r] = -3.0e38f; l_run[r] = 0.f; }

    const int nkb = (qbase + 128) >> 5;
    for (int kbi = 0; kbi < nkb; ++kbi) {
        const int kb = kbi << 5;
        // ---- stage K tile: B-frag for QK^T (cols = keys, K-dim = head dims)
        {
            const int ks  = tid >> 6;
            const int nt  = (tid >> 5) & 1;
            const int hs  = (tid >> 4) & 1;
            const int key = kb + nt * 16 + (tid & 15);
            const float* p = kbuf + ((size_t)key * NKV + kvh) * HD + ks * 32 + hs * 8;
            *(v16bf*)(lK + tid * 16) = cvt16(*(const float4*)(p),
                                             *(const float4*)(p + 4),
                                             *(const float4*)(p + 16),
                                             *(const float4*)(p + 20));
        }
        // ---- stage V tile: B-frag for PV (cols = head dims, K-dim = keys)
        {
            const int hs = (tid >> 4) & 1;
            const int n  = (tid >> 5) * 16 + (tid & 15);
            __bf16* d = lV + tid * 16;
            #pragma unroll
            for (int e = 0; e < 16; ++e) {
                int key = kb + hs * 8 + e + ((e >= 8) ? 8 : 0);
                d[e] = f2bf(vbuf[((size_t)key * NKV + kvh) * HD + n]);
            }
        }
        __syncthreads();

        if (kb <= qbase + wave * 16 + 15) {   // wave-uniform causal skip
            v8f s0 = {}, s1 = {};
            #pragma unroll
            for (int ks = 0; ks < 4; ++ks) {
                v16bf b0 = *(const v16bf*)(lK + ((ks * 2 + 0) * 32 + lane) * 16);
                v16bf b1 = *(const v16bf*)(lK + ((ks * 2 + 1) * 32 + lane) * 16);
                s0 = __builtin_amdgcn_wmma_f32_16x16x32_bf16(
                    false, qf[ks], false, b0, (short)0, s0, false, false);
                s1 = __builtin_amdgcn_wmma_f32_16x16x32_bf16(
                    false, qf[ks], false, b1, (short)0, s1, false, false);
            }
            const int key0 = kb + mr;
            const int key1 = kb + 16 + mr;
            const int hA = (mr >> 3) & 1;        // same for key0/key1
            const int e0 = (mr & 7);
            const int e1 = (mr & 7) + 8;
            #pragma unroll
            for (int r = 0; r < 8; ++r) {
                int qi = qbase + wave * 16 + r + 8 * half;
                float x0 = (key0 <= qi) ? s0[r] : -3.0e38f;
                float x1 = (key1 <= qi) ? s1[r] : -3.0e38f;
                float rm = fmaxf(x0, x1);
                rm = fmaxf(rm, __shfl_xor(rm, 1, 32));
                rm = fmaxf(rm, __shfl_xor(rm, 2, 32));
                rm = fmaxf(rm, __shfl_xor(rm, 4, 32));
                rm = fmaxf(rm, __shfl_xor(rm, 8, 32));
                float mnew = fmaxf(m_run[r], rm);
                float corr = __expf(m_run[r] - mnew);
                float p0 = __expf(x0 - mnew);
                float p1 = __expf(x1 - mnew);
                float rs = p0 + p1;
                rs += __shfl_xor(rs, 1, 32);
                rs += __shfl_xor(rs, 2, 32);
                rs += __shfl_xor(rs, 4, 32);
                rs += __shfl_xor(rs, 8, 32);
                l_run[r] = l_run[r] * corr + rs;
                m_run[r] = mnew;
                #pragma unroll
                for (int j = 0; j < 8; ++j) oacc[j][r] *= corr;
                int m = r + 8 * half;
                lP[wave][(hA * 16 + m) * 16 + e0] = f2bf(p0);
                lP[wave][(hA * 16 + m) * 16 + e1] = f2bf(p1);
            }
            v16bf pf = *(const v16bf*)(lP[wave] + lane * 16);
            #pragma unroll
            for (int j = 0; j < 8; ++j) {
                v16bf vf = *(const v16bf*)(lV + (j * 32 + lane) * 16);
                oacc[j] = __builtin_amdgcn_wmma_f32_16x16x32_bf16(
                    false, pf, false, vf, (short)0, oacc[j], false, false);
            }
        }
        __syncthreads();
    }

    #pragma unroll
    for (int r = 0; r < 8; ++r) {
        float inv = 1.f / l_run[r];
        int row = qbase + wave * 16 + r + 8 * half;
        float* op = out + ((size_t)row * NH + head) * HD + mr;
        #pragma unroll
        for (int j = 0; j < 8; ++j) op[j * 16] = oacc[j][r] * inv;
    }
}

// ---------------------------------------------------------------------------
// Embedding: text token lookup or sum over NCB audio codebooks.
// ---------------------------------------------------------------------------
__global__ __launch_bounds__(256)
void embed_kernel(const int* __restrict__ ids, const float* __restrict__ vocab_emb,
                  const float* __restrict__ audio_emb, float* __restrict__ h) {
    int s  = blockIdx.x;
    int id = ids[s];
    float* out = h + (size_t)s * HDIM;
    if (id > TEXT_VOCAB - 1) {
        int tok = id - TEXT_VOCAB;
        for (int e = threadIdx.x; e < HDIM; e += 256) {
            float acc = 0.f;
            #pragma unroll
            for (int cb = 0; cb < NCB; ++cb)
                acc += audio_emb[(size_t)(tok + cb * CBK) * HDIM + e];
            out[e] = acc;
        }
    } else {
        const float* row = vocab_emb + (size_t)id * HDIM;
        for (int e = threadIdx.x; e < HDIM; e += 256) out[e] = row[e];
    }
}

// ---------------------------------------------------------------------------
// RMSNorm with per-token (text/audio) weight selection.
// ---------------------------------------------------------------------------
__global__ __launch_bounds__(256)
void rmsnorm_kernel(const float* __restrict__ x, const int* __restrict__ ids,
                    const float* __restrict__ wt, const float* __restrict__ wa,
                    float* __restrict__ out) {
    __shared__ float red[256];
    int s = blockIdx.x;
    const float* row = x + (size_t)s * HDIM;
    float ss = 0.f;
    for (int e = threadIdx.x; e < HDIM; e += 256) { float v = row[e]; ss += v * v; }
    red[threadIdx.x] = ss;
    __syncthreads();
    for (int off = 128; off > 0; off >>= 1) {
        if ((int)threadIdx.x < off) red[threadIdx.x] += red[threadIdx.x + off];
        __syncthreads();
    }
    float scale = rsqrtf(red[0] / (float)HDIM + EPSF);
    const float* w = (ids[s] > TEXT_VOCAB - 1) ? wa : wt;
    float* o = out + (size_t)s * HDIM;
    for (int e = threadIdx.x; e < HDIM; e += 256) o[e] = row[e] * scale * w[e];
}

// ---------------------------------------------------------------------------
// RoPE (in place). x shape (SEQ, nheads, HD). Block = 64 threads (j = 0..63).
// ---------------------------------------------------------------------------
__global__ __launch_bounds__(64)
void rope_kernel(float* __restrict__ x, int nheads) {
    int s  = blockIdx.x;
    int hh = blockIdx.y;
    int j  = threadIdx.x;
    float inv = __powf(10000.f, -(float)(2 * j) / (float)HD);
    float ang = (float)s * inv;
    float c  = cosf(ang);
    float sn = sinf(ang);
    float* p = x + ((size_t)s * nheads + hh) * HD;
    float x1 = p[j], x2 = p[j + 64];
    p[j]      = x1 * c - x2 * sn;
    p[j + 64] = x2 * c + x1 * sn;
}

// ---------------------------------------------------------------------------
// Elementwise helpers.
// ---------------------------------------------------------------------------
__global__ __launch_bounds__(256)
void add_inplace_kernel(float* __restrict__ a, const float* __restrict__ b, int n) {
    int i = blockIdx.x * 256 + threadIdx.x;
    if (i < n) a[i] += b[i];
}

__global__ __launch_bounds__(256)
void silu_inplace_kernel(float* __restrict__ x, int n) {
    int i = blockIdx.x * 256 + threadIdx.x;
    if (i < n) { float v = x[i]; x[i] = v / (1.f + __expf(-v)); }
}

__global__ __launch_bounds__(256)
void combine_kernel(float* __restrict__ h, const float* __restrict__ t,
                    const float* __restrict__ a, const int* __restrict__ ids, int n) {
    int i = blockIdx.x * 256 + threadIdx.x;
    if (i < n) {
        int row = i >> 11;
        h[i] += (ids[row] > TEXT_VOCAB - 1) ? a[i] : t[i];
    }
}

// ---------------------------------------------------------------------------
extern "C" void kernel_launch(void* const* d_in, const int* in_sizes, int n_in,
                              void* d_out, int out_size, void* d_ws, size_t ws_size,
                              hipStream_t stream) {
    (void)in_sizes; (void)n_in; (void)out_size; (void)ws_size;

    const int*   ids        = (const int*)  d_in[0];
    const float* vocab_emb  = (const float*)d_in[1];
    const float* audio_emb  = (const float*)d_in[2];
    const float* ln_in_t    = (const float*)d_in[3];
    const float* ln_in_a    = (const float*)d_in[4];
    const float* ln_po_t    = (const float*)d_in[5];
    const float* ln_po_a    = (const float*)d_in[6];
    const float* wq_all     = (const float*)d_in[7];
    const float* wk_all     = (const float*)d_in[8];
    const float* wv_all     = (const float*)d_in[9];
    const float* wo_all     = (const float*)d_in[10];
    const float* wfc_t_all  = (const float*)d_in[11];
    const float* wpr_t_all  = (const float*)d_in[12];
    const float* wfc_a_all  = (const float*)d_in[13];
    const float* wpr_a_all  = (const float*)d_in[14];
    const float* ln_f       = (const float*)d_in[15];

    const size_t SH  = (size_t)SEQ * HDIM;
    const size_t SKV = (size_t)SEQ * NKV * HD;

    float* h    = (float*)d_ws;
    float* hn   = h    + SH;
    float* qb   = hn   + SH;
    float* tb   = qb   + SH;
    float* attn = tb   + SH;
    float* kb   = attn + SH;
    float* vb   = kb   + SKV;
    float* ffb  = vb   + SKV;

    embed_kernel<<<SEQ, 256, 0, stream>>>(ids, vocab_emb, audio_emb, h);

    for (int i = 0; i < NLAYERS; ++i) {
        const float* wq    = wq_all    + (size_t)i * HDIM * (NH  * HD);
        const float* wk    = wk_all    + (size_t)i * HDIM * (NKV * HD);
        const float* wv    = wv_all    + (size_t)i * HDIM * (NKV * HD);
        const float* wo    = wo_all    + (size_t)i * (NH * HD) * HDIM;
        const float* wfc_t = wfc_t_all + (size_t)i * HDIM * FF;
        const float* wpr_t = wpr_t_all + (size_t)i * FF * HDIM;
        const float* wfc_a = wfc_a_all + (size_t)i * HDIM * FF;
        const float* wpr_a = wpr_a_all + (size_t)i * FF * HDIM;

        // ---- attention block ----
        rmsnorm_kernel<<<SEQ, 256, 0, stream>>>(h, ids, ln_in_t + i * HDIM,
                                                ln_in_a + i * HDIM, hn);
        gemm_bf16_wmma<<<dim3((NH * HD) / 128, SEQ / 128), 256, 0, stream>>>(
            hn, wq, qb, SEQ, NH * HD, HDIM);
        gemm_bf16_wmma<<<dim3((NKV * HD) / 128, SEQ / 128), 256, 0, stream>>>(
            hn, wk, kb, SEQ, NKV * HD, HDIM);
        gemm_bf16_wmma<<<dim3((NKV * HD) / 128, SEQ / 128), 256, 0, stream>>>(
            hn, wv, vb, SEQ, NKV * HD, HDIM);
        rope_kernel<<<dim3(SEQ, NH),  64, 0, stream>>>(qb, NH);
        rope_kernel<<<dim3(SEQ, NKV), 64, 0, stream>>>(kb, NKV);
        flash_attn_wmma<<<dim3(SEQ / 128, NH), 256, 0, stream>>>(qb, kb, vb, attn);
        gemm_bf16_wmma<<<dim3(HDIM / 128, SEQ / 128), 256, 0, stream>>>(
            attn, wo, tb, SEQ, HDIM, NH * HD);
        add_inplace_kernel<<<(int)(SH / 256), 256, 0, stream>>>(h, tb, (int)SH);

        // ---- MLP text branch ----
        rmsnorm_kernel<<<SEQ, 256, 0, stream>>>(h, ids, ln_po_t + i * HDIM,
                                                ln_po_t + i * HDIM, hn);
        gemm_bf16_wmma<<<dim3(FF / 128, SEQ / 128), 256, 0, stream>>>(
            hn, wfc_t, ffb, SEQ, FF, HDIM);
        silu_inplace_kernel<<<(SEQ * FF) / 256, 256, 0, stream>>>(ffb, SEQ * FF);
        gemm_bf16_wmma<<<dim3(HDIM / 128, SEQ / 128), 256, 0, stream>>>(
            ffb, wpr_t, tb, SEQ, HDIM, FF);

        // ---- MLP audio branch (audio output reuses attn buffer) ----
        rmsnorm_kernel<<<SEQ, 256, 0, stream>>>(h, ids, ln_po_a + i * HDIM,
                                                ln_po_a + i * HDIM, hn);
        gemm_bf16_wmma<<<dim3(FF / 128, SEQ / 128), 256, 0, stream>>>(
            hn, wfc_a, ffb, SEQ, FF, HDIM);
        silu_inplace_kernel<<<(SEQ * FF) / 256, 256, 0, stream>>>(ffb, SEQ * FF);
        gemm_bf16_wmma<<<dim3(HDIM / 128, SEQ / 128), 256, 0, stream>>>(
            ffb, wpr_a, attn, SEQ, HDIM, FF);

        combine_kernel<<<(int)(SH / 256), 256, 0, stream>>>(h, tb, attn, ids, (int)SH);
    }

    rmsnorm_kernel<<<SEQ, 256, 0, stream>>>(h, ids, ln_f, ln_f, (float*)d_out);
}
